// NeuralOperator_68238440398979
// MI455X (gfx1250) — compile-verified, matrix operand order checked by reading, not run on hardware
//
#include <hip/hip_runtime.h>
#include <math.h>

typedef __attribute__((ext_vector_type(16))) _Float16 v16h;
typedef __attribute__((ext_vector_type(8)))  _Float16 v8h;
typedef __attribute__((ext_vector_type(8)))  float    v8f;

#define BM 64          // batch rows per workgroup
#define NTHREADS 256   // 8 wave32s

// ---- LDS layout (bytes) ----
#define ENC_OFF 0               // enc f16 [64][560]  (later h2 f16 [64][528])
#define ENC_STR 560
#define H2_STR  528
#define H1_OFF  71680           // h1 f16 [64][1040]  (later branch f32 [64][272] + t f16)
#define H1_STR  1040
#define BR_STR  272             // branch/interaction f32 stride (floats)
#define T_OFF   (H1_OFF + 69632) // t f16 [64][272]; later bias-path b f32 [64][128]
#define T_STR   272
#define BB_OFF  T_OFF
#define POS_OFF 204800          // pos f32 [64][4]
#define W_OFF   205824          // sensor weights f32 [64][32]
#define NS_OFF  214016          // next_state staging f32 [64][16]
#define SMEM_BYTES 218112

// ---- workspace layout (halves) ----
#define OFF_BW1 0u              // 544x1024
#define OFF_BW2 557056u         // 1024x512
#define OFF_BW3 1081344u        // 512x256
#define OFF_TW2 1212416u        // 256x256

// Pack f32 weight [K][N] (row-major) into WMMA f16 B fragments.
// Fragment tile (nt, kt): 512 halves = 32 lanes x 16 halves.
// lane p=lane>>4, n = nt*16 + (lane&15); half j: g=j>>3,o=j&7 -> K = kt*32+g*16+p*8+o.
__global__ void pack_w16(const float* __restrict__ src, _Float16* __restrict__ dst,
                         int K, int N) {
    int idx = blockIdx.x * blockDim.x + threadIdx.x;
    if (idx >= K * N) return;
    int KT   = K >> 5;
    int tile = idx >> 9;
    int w    = idx & 511;
    int lane = w >> 4;
    int j    = w & 15;
    int nt   = tile / KT;
    int kt   = tile - nt * KT;
    int p    = lane >> 4;
    int n    = (nt << 4) + (lane & 15);
    int g    = j >> 3, o = j & 7;
    int kk   = (kt << 5) + (g << 4) + (p << 3) + o;
    dst[idx] = (_Float16)src[kk * N + n];
}

// Load 16x32 f16 A fragment from an LDS row-major buffer (strideH halves).
// lane<16: M=lane, K groups {0..7,16..23}; lane>=16: M=lane-16, K {8..15,24..31}.
__device__ __forceinline__ v16h load_a(const _Float16* base, int strideH,
                                       int m0, int k0, int lane) {
    const _Float16* p = base + (m0 + (lane & 15)) * strideH + k0 + ((lane >> 4) << 3);
    v8h lo = *(const v8h*)p;
    v8h hi = *(const v8h*)(p + 16);
    return __builtin_shufflevector(lo, hi, 0,1,2,3,4,5,6,7,8,9,10,11,12,13,14,15);
}

__device__ __forceinline__ v8f wmma32(v16h a, v16h b, v8f c) {
    return __builtin_amdgcn_wmma_f32_16x16x32_f16(false, a, false, b, (short)0, c,
                                                  false, false);
}

__global__ void __launch_bounds__(NTHREADS)
fused_neural_operator(const float* __restrict__ state,
                      const float* __restrict__ action,
                      const float* __restrict__ sens,
                      const float* __restrict__ bb1,
                      const float* __restrict__ bb2,
                      const float* __restrict__ bb3,
                      const float* __restrict__ tw1,
                      const float* __restrict__ tb1,
                      const float* __restrict__ tb2,
                      const float* __restrict__ qw1,
                      const float* __restrict__ qb1,
                      const float* __restrict__ qw2,
                      const float* __restrict__ qb2,
                      const float* __restrict__ pw,
                      const float* __restrict__ pb,
                      const float* __restrict__ resw,
                      const _Float16* __restrict__ wsp,
                      float* __restrict__ out, int B) {
    extern __shared__ unsigned char smem[];
    _Float16* sEnc = (_Float16*)(smem + ENC_OFF);
    _Float16* sH1  = (_Float16*)(smem + H1_OFF);
    _Float16* sH2  = (_Float16*)(smem + ENC_OFF);
    _Float16* sT   = (_Float16*)(smem + T_OFF);
    float*    sBr  = (float*)(smem + H1_OFF);
    float*    sB   = (float*)(smem + BB_OFF);
    float*    sPos = (float*)(smem + POS_OFF);
    float*    sW   = (float*)(smem + W_OFF);
    float*    sNS  = (float*)(smem + NS_OFF);

    const int tid  = threadIdx.x;
    const int wave = tid >> 5;
    const int lane = tid & 31;
    const long long g0 = (long long)blockIdx.x * BM;

    // ---- P0: positions ----
    if (tid < BM * 3) {
        int r = tid / 3, c = tid - r * 3;
        long long g = g0 + r;
        sPos[r * 4 + c] = (g < B) ? state[g * 13 + c] : 0.f;
    }
    __syncthreads();
    // ---- P0b: sensor kernel weights w = exp(-2*dist) ----
    for (int k = 0; k < (BM * 32) / NTHREADS; ++k) {
        int idx = tid + k * NTHREADS;
        int r = idx >> 5, s = idx & 31;
        float dx = sens[s * 3 + 0] - sPos[r * 4 + 0];
        float dy = sens[s * 3 + 1] - sPos[r * 4 + 1];
        float dz = sens[s * 3 + 2] - sPos[r * 4 + 2];
        sW[idx] = expf(-2.f * sqrtf(dx * dx + dy * dy + dz * dz));
    }
    __syncthreads();
    // ---- P1: build enc f16 [64][544] ----
    for (int k = 0; k < (BM * 544) / NTHREADS; ++k) {
        int idx = tid + k * NTHREADS;
        int r = idx / 544, e = idx - r * 544;
        int s = e / 17, j = e - s * 17;
        long long g = g0 + r;
        float base = 0.f;
        if (g < B) base = (j < 13) ? state[g * 13 + j] : action[g * 4 + (j - 13)];
        sEnc[r * ENC_STR + e] = (_Float16)(base * sW[r * 32 + s]);
    }
    __syncthreads();

    // ---- P2: layer 1 [64x544]@[544x1024] -> relu -> h1 f16 ----
    {
        const v16h* bw = (const v16h*)(wsp + OFF_BW1);
        int m = wave & 3;
        for (int i = 0; i < 32; ++i) {
            int nt = (wave >> 2) + (i << 1);
            const v16h* bp = bw + (size_t)(nt * 17) * 32 + lane;
            v8f acc = {};
            for (int kt = 0; kt < 17; ++kt) {
                __builtin_prefetch((const void*)(bp + (kt + 1) * 32), 0, 0);
                v16h a = load_a(sEnc, ENC_STR, m << 4, kt << 5, lane);
                acc = wmma32(a, bp[kt * 32], acc);
            }
            int n = (nt << 4) + (lane & 15);
            float bias = bb1[n];
            for (int r = 0; r < 8; ++r) {
                int row = (m << 4) + r + ((lane >> 4) << 3);
                float v = acc[r] + bias;
                sH1[row * H1_STR + n] = (_Float16)(v > 0.f ? v : 0.f);
            }
        }
    }
    __syncthreads();

    // ---- P3: layer 2 [64x1024]@[1024x512] -> relu -> h2 f16 (reuses enc buf) ----
    {
        const v16h* bw = (const v16h*)(wsp + OFF_BW2);
        int m = wave & 3;
        for (int i = 0; i < 16; ++i) {
            int nt = (wave >> 2) + (i << 1);
            const v16h* bp = bw + (size_t)(nt * 32) * 32 + lane;
            v8f acc = {};
            for (int kt = 0; kt < 32; ++kt) {
                __builtin_prefetch((const void*)(bp + (kt + 1) * 32), 0, 0);
                v16h a = load_a(sH1, H1_STR, m << 4, kt << 5, lane);
                acc = wmma32(a, bp[kt * 32], acc);
            }
            int n = (nt << 4) + (lane & 15);
            float bias = bb2[n];
            for (int r = 0; r < 8; ++r) {
                int row = (m << 4) + r + ((lane >> 4) << 3);
                float v = acc[r] + bias;
                sH2[row * H2_STR + n] = (_Float16)(v > 0.f ? v : 0.f);
            }
        }
    }
    __syncthreads();

    // ---- P4: trunk input t = tanh(pos@tw1+tb1)  AND  layer 3 -> branch f32 ----
    for (int k = 0; k < (BM * 256) / NTHREADS; ++k) {
        int idx = tid + k * NTHREADS;
        int r = idx >> 8, n = idx & 255;
        float a = tb1[n] + sPos[r * 4 + 0] * tw1[n]
                         + sPos[r * 4 + 1] * tw1[256 + n]
                         + sPos[r * 4 + 2] * tw1[512 + n];
        sT[r * T_STR + n] = (_Float16)tanhf(a);
    }
    {
        const v16h* bw = (const v16h*)(wsp + OFF_BW3);
        int m = wave & 3;
        for (int i = 0; i < 8; ++i) {
            int nt = (wave >> 2) + (i << 1);
            const v16h* bp = bw + (size_t)(nt * 16) * 32 + lane;
            v8f acc = {};
            for (int kt = 0; kt < 16; ++kt) {
                v16h a = load_a(sH2, H2_STR, m << 4, kt << 5, lane);
                acc = wmma32(a, bp[kt * 32], acc);
            }
            int n = (nt << 4) + (lane & 15);
            float bias = bb3[n];
            for (int r = 0; r < 8; ++r) {
                int row = (m << 4) + r + ((lane >> 4) << 3);
                sBr[row * BR_STR + n] = acc[r] + bias;
            }
        }
    }
    __syncthreads();

    // ---- P5: trunk GEMM [64x256]@[256x256]; interaction = branch * tanh(.) ----
    {
        const v16h* bw = (const v16h*)(wsp + OFF_TW2);
        int m = wave & 3;
        for (int i = 0; i < 8; ++i) {
            int nt = (wave >> 2) + (i << 1);
            const v16h* bp = bw + (size_t)(nt * 8) * 32 + lane;
            v8f acc = {};
            for (int kt = 0; kt < 8; ++kt) {
                v16h a = load_a(sT, T_STR, m << 4, kt << 5, lane);
                acc = wmma32(a, bp[kt * 32], acc);
            }
            int n = (nt << 4) + (lane & 15);
            float bias = tb2[n];
            for (int r = 0; r < 8; ++r) {
                int row = (m << 4) + r + ((lane >> 4) << 3);
                sBr[row * BR_STR + n] *= tanhf(acc[r] + bias);
            }
        }
    }
    __syncthreads();

    // ---- P6a: bias path b = relu(pos@qw1+qb1)  f32 [64][128] (reuses t buf) ----
    for (int k = 0; k < (BM * 128) / NTHREADS; ++k) {
        int idx = tid + k * NTHREADS;
        int r = idx >> 7, j = idx & 127;
        float a = qb1[j] + sPos[r * 4 + 0] * qw1[j]
                         + sPos[r * 4 + 1] * qw1[128 + j]
                         + sPos[r * 4 + 2] * qw1[256 + j];
        sB[r * 128 + j] = a > 0.f ? a : 0.f;
    }
    __syncthreads();

    // ---- P6b: heads + residual ----
    {
        float rw = resw[0];
        for (int k = 0; k < 4; ++k) {
            int idx = tid + k * NTHREADS;
            if (idx < BM * 13) {
                int r = idx / 13, c = idx - r * 13;
                const float* ip = sBr + r * BR_STR;
                float d = pb[c];
                for (int kk = 0; kk < 256; ++kk) d += ip[kk] * pw[kk * 13 + c];
                const float* bp2 = sB + r * 128;
                float bo = qb2[c];
                for (int j = 0; j < 128; ++j) bo += bp2[j] * qw2[j * 13 + c];
                long long g = g0 + r;
                float st = (g < B) ? state[g * 13 + c] : 0.f;
                sNS[r * 16 + c] = st + rw * (d + bo);
            }
        }
    }
    __syncthreads();

    // ---- P6c: quaternion renorm + store ----
    if (tid < BM) {
        int r = tid;
        long long g = g0 + r;
        if (g < B) {
            float q0 = sNS[r * 16 + 3], q1 = sNS[r * 16 + 4];
            float q2 = sNS[r * 16 + 5], q3 = sNS[r * 16 + 6];
            float qn = sqrtf(q0 * q0 + q1 * q1 + q2 * q2 + q3 * q3) + 1e-8f;
            float inv = 1.f / qn;
            for (int c = 0; c < 13; ++c) {
                float v = sNS[r * 16 + c];
                if (c >= 3 && c < 7) v *= inv;
                out[g * 13 + c] = v;
            }
        }
    }
}

extern "C" void kernel_launch(void* const* d_in, const int* in_sizes, int n_in,
                              void* d_out, int out_size, void* d_ws, size_t ws_size,
                              hipStream_t stream) {
    const float* state  = (const float*)d_in[0];
    const float* action = (const float*)d_in[1];
    const float* sens   = (const float*)d_in[2];
    const float* bw1    = (const float*)d_in[3];
    const float* bb1    = (const float*)d_in[4];
    const float* bw2    = (const float*)d_in[5];
    const float* bb2    = (const float*)d_in[6];
    const float* bw3    = (const float*)d_in[7];
    const float* bb3    = (const float*)d_in[8];
    const float* tw1    = (const float*)d_in[9];
    const float* tb1    = (const float*)d_in[10];
    const float* tw2    = (const float*)d_in[11];
    const float* tb2    = (const float*)d_in[12];
    const float* qw1    = (const float*)d_in[13];
    const float* qb1    = (const float*)d_in[14];
    const float* qw2    = (const float*)d_in[15];
    const float* qb2    = (const float*)d_in[16];
    const float* pw     = (const float*)d_in[17];
    const float* pb     = (const float*)d_in[18];
    const float* resw   = (const float*)d_in[19];
    float* out = (float*)d_out;
    int B = in_sizes[0] / 13;

    _Float16* wsp = (_Float16*)d_ws;

    // Pack the four big weight matrices into WMMA B-fragment layout.
    pack_w16<<<(544 * 1024 + 255) / 256, 256, 0, stream>>>(bw1, wsp + OFF_BW1, 544, 1024);
    pack_w16<<<(1024 * 512 + 255) / 256, 256, 0, stream>>>(bw2, wsp + OFF_BW2, 1024, 512);
    pack_w16<<<(512 * 256 + 255) / 256, 256, 0, stream>>>(bw3, wsp + OFF_BW3, 512, 256);
    pack_w16<<<(256 * 256 + 255) / 256, 256, 0, stream>>>(tw2, wsp + OFF_TW2, 256, 256);

    int grid = (B + BM - 1) / BM;
    fused_neural_operator<<<grid, NTHREADS, SMEM_BYTES, stream>>>(
        state, action, sens, bb1, bb2, bb3, tw1, tb1, tb2,
        qw1, qb1, qw2, qb2, pw, pb, resw, wsp, out, B);
}
